// SpatialGRU_43920335569192
// MI455X (gfx1250) — matrix operand order repaired and checked by reading, not compiled, not run on hardware
//
#include <hip/hip_runtime.h>

// ---------------------------------------------------------------------------
// SpatialGRU on gfx1250 (MI455X).
//  B=256, C=128, L1=L2=32, U=128.  q-width = 512, W cols = 896.
//  Strategy: 63 anti-diagonal wavefront launches; each cell = (B-tile=64) WG
//  doing three bf16 WMMA GEMMs (v_wmma_f32_16x16x32_bf16) with fp32 accum.
//  Weights pre-swizzled once per launch into fragment-ready bf16 (L2-resident).
// ---------------------------------------------------------------------------

typedef unsigned short u16;
typedef __attribute__((ext_vector_type(16))) __bf16 v16bf;
typedef __attribute__((ext_vector_type(8)))  float  v8f;

union Frag { v16bf v; uint4 q[2]; };

__device__ __forceinline__ u16 f32_to_bf16(float f) {
  unsigned x = __float_as_uint(f);
  unsigned r = (x + 0x7fffu + ((x >> 16) & 1u)) >> 16;   // round-to-nearest-even
  return (u16)r;
}
__device__ __forceinline__ float bf16_to_f32(u16 h) {
  return __uint_as_float(((unsigned)h) << 16);
}

// ---- constants ----
#define BB   256      // batch
#define UU   128      // hidden
#define KQ   512      // q width (C + 3U)
#define NW   896      // 7U
#define BT   64       // batch tile per workgroup
#define ENT  32768    // elements per H entry (256*128)

// ---------------------------------------------------------------------------
// Pre-swizzle fp32 (K x N) weight into bf16 fragment-ready layout:
// frag f = kt*(N/16)+nt holds 512 bf16: position lane*16+e  <-
//   src[(kt*32 + ((lane&16)>>1) + (e&7) + ((e&8)<<1))*N + nt*16 + (lane&15)]
// so each lane of a wave reads its 32B with two global_load_b128.
// ---------------------------------------------------------------------------
__global__ void swizzleB_kernel(const float* __restrict__ src,
                                u16* __restrict__ dst, int K, int N) {
  int tid = blockIdx.x * blockDim.x + threadIdx.x;
  if (tid >= K * N) return;
  int f = tid >> 9;          // /512
  int p = tid & 511;
  int lane = p >> 4, e = p & 15;
  int ntiles = N >> 4;
  int kt = f / ntiles, nt = f - kt * ntiles;
  int k = kt * 32 + ((lane & 16) >> 1) + (e & 7) + ((e & 8) << 1);
  int n = nt * 16 + (lane & 15);
  dst[tid] = f32_to_bf16(src[(size_t)k * N + n]);
}

// inputs (B,C,L1,L2) fp32  ->  s_bf (L1,L2,B,C) bf16
__global__ void convS_kernel(const float* __restrict__ in, u16* __restrict__ out) {
  size_t tid = (size_t)blockIdx.x * 256 + threadIdx.x;   // 2^25 total
  int c  = (int)(tid & 127);
  int b  = (int)((tid >> 7) & 255);
  int l2 = (int)((tid >> 15) & 31);
  int l1 = (int)((tid >> 20) & 31);
  out[tid] = f32_to_bf16(in[(((size_t)b * 128 + c) * 32 + l1) * 32 + l2]);
}

// Zero H row 0 (33 entries) and H[r][0] for r=0..32 (boundary h == 0).
__global__ void initH_kernel(u16* __restrict__ H) {
  size_t tid = (size_t)blockIdx.x * 256 + threadIdx.x;
  const size_t R0 = (size_t)33 * ENT;
  if (tid < R0) {
    H[tid] = 0;                                   // row 0, all 33 entries
  } else if (tid < 2 * R0) {
    size_t t = tid - R0;
    size_t r = t / ENT, o = t % ENT;
    H[(r * 33) * ENT + o] = 0;                    // column-0 entry of each row
  }
}

__global__ void finalOut_kernel(const u16* __restrict__ H, float* __restrict__ out) {
  int tid = blockIdx.x * 256 + threadIdx.x;
  if (tid < ENT)
    out[tid] = bf16_to_f32(H[(size_t)(32 * 33 + 32) * ENT + tid]);
}

// ---------------------------------------------------------------------------
// One GRU cell, B-tile of 64 rows.  8 waves (wave32), 256 threads.
// Phase A: q(64x512) @ W(512x896) -> r (hard-sigmoid, bf16 LDS) + z (fp32 LDS)
// Phase B: rabuf <- r * [h_left,h_top,h_diag]  (in place, bf16)
// Phase C: hij_ = tanh(rabuf@Umat + s@w_ij + b) ; softmax gates ; write H
// ---------------------------------------------------------------------------
__global__ __launch_bounds__(256)
void cell_kernel(int d,
                 const u16* __restrict__ sbf,
                 u16* __restrict__ H,
                 const u16* __restrict__ Wsw,
                 const u16* __restrict__ Usw,
                 const u16* __restrict__ Wijsw,
                 const float* __restrict__ bias) {
  extern __shared__ char smem[];
  float* zbuf  = (float*)smem;                    // [64][512] fp32 (128 KB)
  u16*   rabuf = (u16*)(smem + 64 * 512 * 4);     // [64][384] bf16 ( 48 KB)

  const int i0 = (d > 31) ? (d - 31) : 0;
  const int i  = i0 + blockIdx.y;
  const int j  = d - i;
  const int bbase = blockIdx.x * BT;

  const u16* hTop  = H + ((size_t)i       * 33 + (j + 1)) * ENT;
  const u16* hLeft = H + ((size_t)(i + 1) * 33 +  j     ) * ENT;
  const u16* hDiag = H + ((size_t)i       * 33 +  j     ) * ENT;
  const u16* sij   = sbf + (size_t)(i * 32 + j) * ENT;
  u16*       hOut  = H + ((size_t)(i + 1) * 33 + (j + 1)) * ENT;

  const int tid  = threadIdx.x;
  const int wave = tid >> 5;
  const int lane = tid & 31;
  const int mt   = wave & 3;        // M tile (16 rows) within B-tile
  const int nh   = wave >> 2;       // N half for phase A
  const int mrow = lane & 15;       // A-fragment row within tile
  const int hiA  = lane >> 4;       // A-fragment K-chunk selector

  // ---- load all 16 A-fragments of q into registers (reused for 28 N-tiles)
  const u16* qsrc[4] = { hTop, hLeft, hDiag, sij };
  Frag a[16];
#pragma unroll
  for (int kt = 0; kt < 16; ++kt) {
    const u16* p = qsrc[kt >> 2]
                 + (size_t)(bbase + mt * 16 + mrow) * UU
                 + (kt & 3) * 32 + hiA * 8;
    a[kt].q[0] = *(const uint4*)(p);        // K +0..7  (or +8..15)
    a[kt].q[1] = *(const uint4*)(p + 16);   // K +16..23 (or +24..31)
  }

  // ---- Phase A: z/r GEMM -----------------------------------------------
  for (int t = 0; t < 28; ++t) {
    const int nt = nh * 28 + t;
    v8f acc = {};
    const u16* wp = Wsw + (size_t)nt * 512 + (size_t)lane * 16;
#pragma unroll
    for (int kt = 0; kt < 16; ++kt) {
      Frag b;
      const u16* bp = wp + (size_t)kt * (NW / 16) * 512;
      b.q[0] = *(const uint4*)(bp);
      b.q[1] = *(const uint4*)(bp + 8);
      acc = __builtin_amdgcn_wmma_f32_16x16x32_bf16(
              false, a[kt].v, false, b.v, (short)0, acc, false, false);
    }
    const int ncol  = nt * 16 + (lane & 15);
    const int mbase = mt * 16 + hiA * 8;
    const float bs  = bias[ncol];
#pragma unroll
    for (int v = 0; v < 8; ++v) {
      const int m = mbase + v;
      float val = acc[v] + bs;
      if (ncol < 384) {         // reset gates: hard sigmoid, keep bf16
        float r = fminf(fmaxf(0.2f * val + 0.5f, 0.0f), 1.0f);
        rabuf[m * 384 + ncol] = f32_to_bf16(r);
      } else {                  // update gates: raw logits for softmax
        zbuf[m * 512 + (ncol - 384)] = val;
      }
    }
  }
  __syncthreads();

  // ---- Phase B: rabuf <- r * [h_left, h_top, h_diag] (Umat operand order)
  {
    const u16* hsrc[3] = { hLeft, hTop, hDiag };
    for (int idx = tid; idx < 64 * 384; idx += 256) {
      const int m = idx / 384, k = idx - m * 384;
      float r = bf16_to_f32(rabuf[idx]);
      float h = bf16_to_f32(hsrc[k >> 7][(size_t)(bbase + m) * UU + (k & 127)]);
      rabuf[idx] = f32_to_bf16(r * h);
    }
  }
  __syncthreads();

  // ---- Phase C: hij_ GEMM + gate combine --------------------------------
  const int ntg = (wave >> 2) * 4;                 // 4 N-tiles of U per wave
  for (int t2 = 0; t2 < 4; ++t2) {
    const int nt = ntg + t2;
    v8f acc = {};
#pragma unroll
    for (int kt = 0; kt < 12; ++kt) {              // rabuf(64x384) @ Umat
      Frag af, bf;
      const u16* ap = rabuf + (mt * 16 + mrow) * 384 + kt * 32 + hiA * 8;
      af.q[0] = *(const uint4*)(ap);
      af.q[1] = *(const uint4*)(ap + 16);
      const u16* bp = Usw + (size_t)(kt * 8 + nt) * 512 + (size_t)lane * 16;
      bf.q[0] = *(const uint4*)(bp);
      bf.q[1] = *(const uint4*)(bp + 8);
      acc = __builtin_amdgcn_wmma_f32_16x16x32_bf16(
              false, af.v, false, bf.v, (short)0, acc, false, false);
    }
#pragma unroll
    for (int kt = 0; kt < 4; ++kt) {               // s_ij(64x128) @ w_ij
      Frag af, bf;
      const u16* ap = sij + (size_t)(bbase + mt * 16 + mrow) * UU
                    + kt * 32 + hiA * 8;
      af.q[0] = *(const uint4*)(ap);
      af.q[1] = *(const uint4*)(ap + 16);
      const u16* bp = Wijsw + (size_t)(kt * 8 + nt) * 512 + (size_t)lane * 16;
      bf.q[0] = *(const uint4*)(bp);
      bf.q[1] = *(const uint4*)(bp + 8);
      acc = __builtin_amdgcn_wmma_f32_16x16x32_bf16(
              false, af.v, false, bf.v, (short)0, acc, false, false);
    }
    const int u     = nt * 16 + (lane & 15);
    const int mbase = mt * 16 + hiA * 8;
    const float bij = bias[896 + u];
#pragma unroll
    for (int v = 0; v < 8; ++v) {
      const int m = mbase + v;
      float hpre = tanhf(acc[v] + bij);
      float z0 = zbuf[m * 512 +       u];          // zi -> hij_
      float z1 = zbuf[m * 512 + 128 + u];          // zl -> h_left
      float z2 = zbuf[m * 512 + 256 + u];          // zt -> h_top
      float z3 = zbuf[m * 512 + 384 + u];          // zd -> h_diag
      float zm = fmaxf(fmaxf(z0, z1), fmaxf(z2, z3));
      float e0 = __expf(z0 - zm), e1 = __expf(z1 - zm);
      float e2 = __expf(z2 - zm), e3 = __expf(z3 - zm);
      float inv = 1.0f / (e0 + e1 + e2 + e3);
      size_t go = (size_t)(bbase + m) * UU + u;
      float hl = bf16_to_f32(hLeft[go]);
      float ht = bf16_to_f32(hTop[go]);
      float hd = bf16_to_f32(hDiag[go]);
      float hij = (e1 * hl + e2 * ht + e3 * hd + e0 * hpre) * inv;
      hOut[go] = f32_to_bf16(hij);
    }
  }
}

// ---------------------------------------------------------------------------
extern "C" void kernel_launch(void* const* d_in, const int* in_sizes, int n_in,
                              void* d_out, int out_size, void* d_ws, size_t ws_size,
                              hipStream_t stream) {
  const float* inputs = (const float*)d_in[0];   // (256,128,32,32)
  const float* W      = (const float*)d_in[1];   // (512,896)
  const float* Umat   = (const float*)d_in[2];   // (384,128)
  const float* bias   = (const float*)d_in[3];   // (1024,)
  const float* w_ij   = (const float*)d_in[4];   // (128,128)

  char* ws = (char*)d_ws;
  u16* Wsw   = (u16*)(ws);                       //   917504 B
  u16* Usw   = (u16*)(ws + 917504);              //    98304 B
  u16* Wijsw = (u16*)(ws + 1015808);             //    32768 B
  u16* sbf   = (u16*)(ws + 1048576);             // 67108864 B
  u16* H     = (u16*)(ws + 68157440);            // 71368704 B  (33x33 entries)

  // one-time (per launch) weight swizzle + input transpose/convert + H init
  swizzleB_kernel<<<(KQ * NW + 255) / 256, 256, 0, stream>>>(W, Wsw, KQ, NW);
  swizzleB_kernel<<<(384 * 128 + 255) / 256, 256, 0, stream>>>(Umat, Usw, 384, 128);
  swizzleB_kernel<<<(128 * 128 + 255) / 256, 256, 0, stream>>>(w_ij, Wijsw, 128, 128);
  convS_kernel<<<(1u << 25) / 256, 256, 0, stream>>>(inputs, sbf);
  {
    size_t total = 2ull * 33 * ENT;
    initH_kernel<<<(unsigned)((total + 255) / 256), 256, 0, stream>>>(H);
  }

  // anti-diagonal wavefront: stream order supplies the dependency
  const size_t shmem = 64 * 512 * 4 + 64 * 384 * 2;   // 180224 B
  for (int d = 0; d <= 62; ++d) {
    int ilo = d > 31 ? d - 31 : 0;
    int ihi = d < 31 ? d : 31;
    dim3 grid(4, (unsigned)(ihi - ilo + 1));
    cell_kernel<<<grid, 256, shmem, stream>>>(d, sbf, H, Wsw, Usw, Wijsw, bias);
  }

  finalOut_kernel<<<(ENT + 255) / 256, 256, 0, stream>>>(H, (float*)d_out);
}
